// Model_40630390620492
// MI455X (gfx1250) — compile-verified, matrix operand order checked by reading, not run on hardware
//
#include <hip/hip_runtime.h>
#include <cstdint>
#include <cstddef>

// ---------------------------------------------------------------------------
// GPT causal self-attention block for gfx1250 (MI455X), wave32 + WMMA bf16.
//   B=8, T=1024, C=768, H=8, D=96.
//   Pipeline: cvt(f32->bf16) -> QKV GEMM (wmma) -> flash attention (wmma)
//             -> output projection GEMM (wmma, fp32 out).
// All LDS tiles padded so every WMMA fragment is two 16B-aligned runs:
// fragments load as 2x ds_load_b128 / global_load_b128.
// ---------------------------------------------------------------------------

typedef __bf16 bf16;
typedef __attribute__((ext_vector_type(8)))  __bf16 bf16x8;
typedef __attribute__((ext_vector_type(16))) __bf16 bf16x16;
typedef __attribute__((ext_vector_type(8)))  float  f32x8;

union U32B  { uint32_t u; bf16 h[2]; };
union U128B { uint4 q;    bf16 h[8]; };

#define CDIM   768
#define TDIM   1024
#define BDIM   8
#define HEADS  8
#define HD     96
#define QKV_SCALE 0.1020620726159658f   // 1/sqrt(96)

#define LDSK   40    // GEMM LDS tile stride (elements): 80B rows, 16B-aligned,
                     // frag reads hit banks r*20 mod 64 -> conflict-free
#define KS_STRIDE 104  // 208B rows (13*16)
#define VT_STRIDE 72   // 144B rows (9*16)
#define PS_STRIDE 72

// ---------------------------------------------------------------------------
// Fragment loader for V_WMMA_F32_16X16X32_BF16 (wave32).
// A-matrix 16x32 (MxK) layout (ISA 7.12.2): lane L holds row m=L&15;
//   VGPRs 0..3 -> k = 8*(L>>4) + 0..7   (contiguous 16B)
//   VGPRs 4..7 -> k = 16 + 8*(L>>4) + 0..7 (contiguous 16B)
// => exactly two b128 loads per fragment. B fragment = same layout on B^T
// (rows = N, contiguous in K). Caller guarantees 16B alignment of base and
// 16B-multiple row stride.
// ---------------------------------------------------------------------------
__device__ inline bf16x16 load_frag_nk(const bf16* base, int stride) {
  const int lane = threadIdx.x & 31;
  const bf16* p = base + (lane & 15) * stride + ((lane >> 4) << 3);
  bf16x8 lo = *(const bf16x8*)(p);
  bf16x8 hi = *(const bf16x8*)(p + 16);
  return __builtin_shufflevector(lo, hi, 0, 1, 2, 3, 4, 5, 6, 7,
                                 8, 9, 10, 11, 12, 13, 14, 15);
}

__device__ inline f32x8 wmma_bf16(bf16x16 a, bf16x16 b, f32x8 c) {
  // D = A(16x32) * B(32x16) + C, f32 accumulate.
  return __builtin_amdgcn_wmma_f32_16x16x32_bf16(
      false, a, false, b, (short)0, c, false, false);
}

// ---------------------------------------------------------------------------
// fp32 -> bf16 conversion, 2 elements/thread packed into one b32 store.
// ---------------------------------------------------------------------------
__global__ __launch_bounds__(256) void cvt_f32_bf16(const float2* __restrict__ in,
                                                    uint32_t* __restrict__ out,
                                                    int n2) {
  for (int i = blockIdx.x * 256 + threadIdx.x; i < n2; i += gridDim.x * 256) {
    float2 v = in[i];
    U32B u;
    u.h[0] = (bf16)v.x;
    u.h[1] = (bf16)v.y;
    out[i] = u.u;
  }
}

// ---------------------------------------------------------------------------
// Shared 128x128 GEMM mainloop: C(128x128) += A(128xK) * B(128xK)^T
// A rows = M, B rows = N, both contiguous in K.  BK=32, 8 waves as 2x4,
// each wave owns a 64x32 tile = 4x2 WMMA accumulators.
// Register double-buffer: next K-slice is prefetched with global_load_b128
// while the current slice runs ds_load_b128 + 8 WMMAs.
// ---------------------------------------------------------------------------
__device__ inline void gemm_mainloop(const bf16* __restrict__ A,
                                     const bf16* __restrict__ Bw,
                                     int Kdim, int mBase, int nBase,
                                     bf16* As, bf16* Bs, f32x8 acc[4][2]) {
  const int tid = threadIdx.x;
  const int w = tid >> 5, wm = w >> 2, wn = w & 3;
  const int ldRow = tid >> 2;          // 0..63
  const int ldCol = (tid & 3) << 3;    // 0,8,16,24 elements (16B steps)

  uint4 ra[2], rb[2];
#pragma unroll
  for (int it = 0; it < 2; ++it) {
    const int r = ldRow + it * 64;
    ra[it] = *(const uint4*)(A + (size_t)(mBase + r) * Kdim + ldCol);
    rb[it] = *(const uint4*)(Bw + (size_t)(nBase + r) * Kdim + ldCol);
  }

  for (int kk = 0; kk < Kdim; kk += 32) {
#pragma unroll
    for (int it = 0; it < 2; ++it) {
      const int r = ldRow + it * 64;
      *(uint4*)(As + r * LDSK + ldCol) = ra[it];
      *(uint4*)(Bs + r * LDSK + ldCol) = rb[it];
    }
    __syncthreads();

    if (kk + 32 < Kdim) {              // prefetch next slice during compute
#pragma unroll
      for (int it = 0; it < 2; ++it) {
        const int r = ldRow + it * 64;
        ra[it] = *(const uint4*)(A + (size_t)(mBase + r) * Kdim + kk + 32 + ldCol);
        rb[it] = *(const uint4*)(Bw + (size_t)(nBase + r) * Kdim + kk + 32 + ldCol);
      }
    }

    bf16x16 af[4], bfr[2];
#pragma unroll
    for (int i = 0; i < 4; ++i)
      af[i] = load_frag_nk(As + (wm * 64 + i * 16) * LDSK, LDSK);
#pragma unroll
    for (int j = 0; j < 2; ++j)
      bfr[j] = load_frag_nk(Bs + (wn * 32 + j * 16) * LDSK, LDSK);
#pragma unroll
    for (int i = 0; i < 4; ++i)
#pragma unroll
      for (int j = 0; j < 2; ++j)
        acc[i][j] = wmma_bf16(af[i], bfr[j], acc[i][j]);
    __syncthreads();
  }
}

// ---------------------------------------------------------------------------
// Kernel 2: QKV projection. qkv[m, f] = x[m,:] . Wattn[f,:] + bias[f]
// Epilogue scatters into Q/K/V in (B,H,T,D) bf16 layout; Q pre-scaled.
// ---------------------------------------------------------------------------
__global__ __launch_bounds__(256) void gemm_qkv_kernel(
    const bf16* __restrict__ xb, const bf16* __restrict__ wb,
    const float* __restrict__ bias,
    bf16* __restrict__ Q, bf16* __restrict__ Kq, bf16* __restrict__ V) {
  __shared__ bf16 As[128 * LDSK];
  __shared__ bf16 Bs[128 * LDSK];

  const f32x8 zero = {0.f, 0.f, 0.f, 0.f, 0.f, 0.f, 0.f, 0.f};
  f32x8 acc[4][2];
#pragma unroll
  for (int i = 0; i < 4; ++i)
#pragma unroll
    for (int j = 0; j < 2; ++j) acc[i][j] = zero;

  const int mBase = blockIdx.y * 128, nBase = blockIdx.x * 128;
  gemm_mainloop(xb, wb, CDIM, mBase, nBase, As, Bs, acc);

  const int lane = threadIdx.x & 31, hsel = lane >> 4, lanen = lane & 15;
  const int w = threadIdx.x >> 5, wm = w >> 2, wn = w & 3;
#pragma unroll
  for (int j = 0; j < 2; ++j) {
    const int n = nBase + wn * 32 + j * 16 + lanen;      // 0..2303 (wave-uniform /768)
    const float bv = bias[n];
    const int which = n / CDIM;            // 0=q 1=k 2=v
    const int cc = n - which * CDIM;
    const int h = cc / HD, d = cc - h * HD;
    bf16* dst = (which == 0) ? Q : (which == 1) ? Kq : V;
    const float sc = (which == 0) ? QKV_SCALE : 1.0f;
#pragma unroll
    for (int i = 0; i < 4; ++i)
#pragma unroll
      for (int r = 0; r < 8; ++r) {
        const int m = mBase + wm * 64 + i * 16 + r + 8 * hsel;
        const int b = m >> 10, t = m & 1023;
        const float v = (acc[i][j][r] + bv) * sc;
        dst[((size_t)(b * HEADS + h) * TDIM + t) * HD + d] = (bf16)v;
      }
  }
}

// ---------------------------------------------------------------------------
// Kernel 3: causal flash attention, one (b,h) x 64 query rows per workgroup.
// 4 waves x 16 q-rows. Key blocks of 64 staged in LDS (K row-major, V
// transposed). Online softmax over C-fragment layout, P via per-wave LDS
// scratch to reach A-fragment layout.
// ---------------------------------------------------------------------------
__global__ __launch_bounds__(128) void attn_kernel(
    const bf16* __restrict__ Q, const bf16* __restrict__ Kq,
    const bf16* __restrict__ V, bf16* __restrict__ Y) {
  __shared__ bf16 Ks[64 * KS_STRIDE];        // [key][d]
  __shared__ bf16 Vt[96 * VT_STRIDE];        // [d][key], transposed
  __shared__ bf16 Ps[4][16 * PS_STRIDE];     // per-wave P scratch [q][key]

  const int tid = threadIdx.x;
  const int w = tid >> 5;
  const int lane = tid & 31, hsel = lane >> 4, lanen = lane & 15;
  const int bh = blockIdx.y;
  const int bb = bh >> 3, hh = bh & 7;
  const int qw = blockIdx.x * 64 + w * 16;       // this wave's first q row

  const bf16* Qh = Q  + (size_t)bh * TDIM * HD;
  const bf16* Kh = Kq + (size_t)bh * TDIM * HD;
  const bf16* Vh = V  + (size_t)bh * TDIM * HD;

  // Q fragments live in registers for the whole key loop (16x96 = 3 frags,
  // loaded straight from global: rows are 192B, all offsets 16B-aligned).
  bf16x16 aq[3];
#pragma unroll
  for (int kq = 0; kq < 3; ++kq)
    aq[kq] = load_frag_nk(Qh + (size_t)qw * HD + kq * 32, HD);

  const f32x8 zero = {0.f, 0.f, 0.f, 0.f, 0.f, 0.f, 0.f, 0.f};
  f32x8 o[6];
#pragma unroll
  for (int jj = 0; jj < 6; ++jj) o[jj] = zero;
  float mrow[8], lrow[8];
#pragma unroll
  for (int r = 0; r < 8; ++r) { mrow[r] = -1e30f; lrow[r] = 0.f; }

  bf16* Psw = &Ps[w][0];
  const int nblocks = blockIdx.x + 1;            // causal: keys <= qbase+63

  for (int blk = 0; blk < nblocks; ++blk) {
    const int kb = blk * 64;
    // Cooperative stage of 64-key block with b128 loads:
    // 64 rows x 6 chunks of 8 bf16.  K row-major; V transposed on store.
    for (int idx = tid; idx < 64 * 6; idx += 128) {
      const int row = idx / 6, c = idx % 6;
      *(uint4*)(Ks + row * KS_STRIDE + c * 8) =
          *(const uint4*)(Kh + (size_t)(kb + row) * HD + c * 8);
      U128B u;
      u.q = *(const uint4*)(Vh + (size_t)(kb + row) * HD + c * 8);
#pragma unroll
      for (int jx = 0; jx < 8; ++jx)
        Vt[(c * 8 + jx) * VT_STRIDE + row] = u.h[jx];
    }
    __syncthreads();

    if (kb <= qw + 15) {                         // wave has unmasked keys here
      // S = Q K^T for 4 chunks of 16 keys.
      f32x8 s[4];
#pragma unroll
      for (int c = 0; c < 4; ++c) {
        s[c] = zero;
#pragma unroll
        for (int kq = 0; kq < 3; ++kq)
          s[c] = wmma_bf16(aq[kq],
                           load_frag_nk(Ks + (c * 16) * KS_STRIDE + kq * 32,
                                        KS_STRIDE),
                           s[c]);
      }
      // Causal mask (C-layout: row = r + 8*hsel, col = lanen).
#pragma unroll
      for (int c = 0; c < 4; ++c) {
        const int colg = kb + c * 16 + lanen;
#pragma unroll
        for (int r = 0; r < 8; ++r)
          if (colg > qw + r + 8 * hsel) s[c][r] = -1e30f;
      }
      // Online softmax: per-row reduce across the 16 lanes of this half-wave.
#pragma unroll
      for (int r = 0; r < 8; ++r) {
        float bm = fmaxf(fmaxf(s[0][r], s[1][r]), fmaxf(s[2][r], s[3][r]));
#pragma unroll
        for (int off = 1; off < 16; off <<= 1)
          bm = fmaxf(bm, __shfl_xor(bm, off, 32));
        const float nm = fmaxf(mrow[r], bm);
        const float sc = __expf(mrow[r] - nm);
        mrow[r] = nm;
        float bs = 0.f;
#pragma unroll
        for (int c = 0; c < 4; ++c) { s[c][r] = __expf(s[c][r] - nm); bs += s[c][r]; }
#pragma unroll
        for (int off = 1; off < 16; off <<= 1)
          bs += __shfl_xor(bs, off, 32);
        lrow[r] = lrow[r] * sc + bs;
#pragma unroll
        for (int jj = 0; jj < 6; ++jj) o[jj][r] *= sc;
      }
      // P (C-layout) -> per-wave LDS -> A-fragment layout.
#pragma unroll
      for (int c = 0; c < 4; ++c)
#pragma unroll
        for (int r = 0; r < 8; ++r)
          Psw[(r + 8 * hsel) * PS_STRIDE + c * 16 + lanen] = (bf16)s[c][r];
      // O += P * V  (2 k-steps of 32 keys x 6 d-tiles).
#pragma unroll
      for (int ka = 0; ka < 2; ++ka) {
        const bf16x16 ap = load_frag_nk(Psw + ka * 32, PS_STRIDE);
#pragma unroll
        for (int jj = 0; jj < 6; ++jj)
          o[jj] = wmma_bf16(ap,
                            load_frag_nk(Vt + (jj * 16) * VT_STRIDE + ka * 32,
                                         VT_STRIDE),
                            o[jj]);
      }
    }
    __syncthreads();
  }

  // Epilogue: normalize and scatter into y (B,T,C) bf16 workspace.
#pragma unroll
  for (int jj = 0; jj < 6; ++jj)
#pragma unroll
    for (int r = 0; r < 8; ++r) {
      const int t = qw + r + 8 * hsel;
      const int d = jj * 16 + lanen;
      const float v = o[jj][r] / lrow[r];
      Y[((size_t)(bb * TDIM + t)) * CDIM + hh * HD + d] = (bf16)v;
    }
}

// ---------------------------------------------------------------------------
// Kernel 4: output projection. out[m, f] = y[m,:] . Wproj[f,:] + bias[f]
// ---------------------------------------------------------------------------
__global__ __launch_bounds__(256) void gemm_proj_kernel(
    const bf16* __restrict__ yb, const bf16* __restrict__ wb,
    const float* __restrict__ bias, float* __restrict__ out) {
  __shared__ bf16 As[128 * LDSK];
  __shared__ bf16 Bs[128 * LDSK];

  const f32x8 zero = {0.f, 0.f, 0.f, 0.f, 0.f, 0.f, 0.f, 0.f};
  f32x8 acc[4][2];
#pragma unroll
  for (int i = 0; i < 4; ++i)
#pragma unroll
    for (int j = 0; j < 2; ++j) acc[i][j] = zero;

  const int mBase = blockIdx.y * 128, nBase = blockIdx.x * 128;
  gemm_mainloop(yb, wb, CDIM, mBase, nBase, As, Bs, acc);

  const int lane = threadIdx.x & 31, hsel = lane >> 4, lanen = lane & 15;
  const int w = threadIdx.x >> 5, wm = w >> 2, wn = w & 3;
#pragma unroll
  for (int j = 0; j < 2; ++j) {
    const int n = nBase + wn * 32 + j * 16 + lanen;
    const float bv = bias[n];
#pragma unroll
    for (int i = 0; i < 4; ++i)
#pragma unroll
      for (int r = 0; r < 8; ++r) {
        const int m = mBase + wm * 64 + i * 16 + r + 8 * hsel;
        out[(size_t)m * CDIM + n] = acc[i][j][r] + bv;
      }
  }
}

// ---------------------------------------------------------------------------
// Launch: workspace layout (bytes, all 256-aligned)
//   xb      @ 0         8192*768  bf16
//   wqkv_b  @ 12582912  2304*768  bf16
//   wproj_b @ 16121856   768*768  bf16
//   Q       @ 17301504  (B,H,T,D) bf16   (pre-scaled by 1/sqrt(D))
//   K       @ 29884416  (B,H,T,D) bf16
//   V       @ 42467328  (B,H,T,D) bf16
//   Y       @ 55050240  (B,T,C)   bf16
// Total ~67.6 MB.
// ---------------------------------------------------------------------------
extern "C" void kernel_launch(void* const* d_in, const int* in_sizes, int n_in,
                              void* d_out, int out_size, void* d_ws, size_t ws_size,
                              hipStream_t stream) {
  const float* x     = (const float*)d_in[0];
  const float* wattn = (const float*)d_in[1];
  const float* battn = (const float*)d_in[2];
  const float* wproj = (const float*)d_in[3];
  const float* bproj = (const float*)d_in[4];
  float* out = (float*)d_out;

  char* ws = (char*)d_ws;
  bf16* xb     = (bf16*)(ws + 0);
  bf16* wqkvb  = (bf16*)(ws + 12582912);
  bf16* wprojb = (bf16*)(ws + 16121856);
  bf16* Q      = (bf16*)(ws + 17301504);
  bf16* Kq     = (bf16*)(ws + 29884416);
  bf16* V      = (bf16*)(ws + 42467328);
  bf16* Y      = (bf16*)(ws + 55050240);

  const int n_x2  = BDIM * TDIM * CDIM / 2;     // float2 pairs
  const int n_wa2 = 3 * CDIM * CDIM / 2;
  const int n_wp2 = CDIM * CDIM / 2;

  cvt_f32_bf16<<<(n_x2  + 255) / 256, 256, 0, stream>>>(
      (const float2*)x, (uint32_t*)xb, n_x2);
  cvt_f32_bf16<<<(n_wa2 + 255) / 256, 256, 0, stream>>>(
      (const float2*)wattn, (uint32_t*)wqkvb, n_wa2);
  cvt_f32_bf16<<<(n_wp2 + 255) / 256, 256, 0, stream>>>(
      (const float2*)wproj, (uint32_t*)wprojb, n_wp2);

  // QKV GEMM: M=8192, N=2304, K=768 -> grid (18, 64)
  gemm_qkv_kernel<<<dim3(3 * CDIM / 128, BDIM * TDIM / 128), 256, 0, stream>>>(
      xb, wqkvb, battn, Q, Kq, V);

  // Flash attention: grid (T/64, B*H) = (16, 64), 128 threads (4 waves)
  attn_kernel<<<dim3(TDIM / 64, BDIM * HEADS), 128, 0, stream>>>(Q, Kq, V, Y);

  // Output projection: M=8192, N=768, K=768 -> grid (6, 64)
  gemm_proj_kernel<<<dim3(CDIM / 128, BDIM * TDIM / 128), 256, 0, stream>>>(
      Y, wprojb, bproj, out);
}
